// ReadoutFunction_29317446762810
// MI455X (gfx1250) — compile-verified, matrix operand order checked by reading, not run on hardware
//
#include <hip/hip_runtime.h>
#include <hip/hip_bf16.h>

// ---------------------------------------------------------------------------
// Attention-weighted graph readout for MI455X (gfx1250, wave32).
//   x:[1e6,256] f32, W1:[256,128], b1:[128], W2:[128,1], b2:[1],
//   batch:[1e6] int64 (sorted graph ids in [0,4096))
// out:[4096,256] f32
//
// Design (see analysis): x streamed twice (HBM floor ~88us @ 23.3TB/s);
// the 65.5 GFLOP x@W1 GEMM runs on v_wmma_f32_16x16x32_f16 (f16 in, f32 acc)
// so compute stays far under the bandwidth floor. Weighted-sum pass is pure
// coalesced streaming, one block per graph, no atomics.
// ---------------------------------------------------------------------------

typedef __attribute__((ext_vector_type(16))) _Float16 v16h;
typedef __attribute__((ext_vector_type(8)))  float    v8f;

#define NUM_NODES 1000000
#define HIDDEN 256
#define RH 128
#define NUM_GRAPHS 4096
#define NUM_TILES (NUM_NODES / 16)   // 62500 exactly

// ---------------------------------------------------------------------------
// W1 [256,128] f32 row-major  ->  W1T [128][256] f16 (col-major of W1), so a
// B-fragment K-pair (k,k+1) for a fixed output column n is contiguous.
// ---------------------------------------------------------------------------
__global__ void prep_w1t_kernel(const float* __restrict__ W1,
                                _Float16* __restrict__ w1t) {
  int i = blockIdx.x * blockDim.x + threadIdx.x;   // 128*256 threads
  if (i < RH * HIDDEN) {
    int n = i >> 8;        // 0..127
    int k = i & 255;       // 0..255
    w1t[i] = (_Float16)W1[k * RH + n];
  }
}

__global__ void init_stats_kernel(float* __restrict__ gmax,
                                  float* __restrict__ gsum) {
  int i = blockIdx.x * blockDim.x + threadIdx.x;
  if (i < NUM_GRAPHS) { gmax[i] = 0.0f; gsum[i] = 0.0f; }
}

// ---------------------------------------------------------------------------
// Per-node attention logit: w = sigmoid(relu(x@W1 + b1)@W2 + b2).
// One wave32 per 16-node tile. A/B fragments packed per CDNA5 ISA 7.12.2:
//   16-bit A 16x32: lane l<16 -> M=l, VGPR p holds K pairs
//     {0,1},{2,3},{4,5},{6,7},{16,17},{18,19},{20,21},{22,23}; lanes 16..31
//     hold the +8 K-offset halves. B mirrors with N in place of M.
//   f32 C/D 16x16: VGPR r = row r (lanes 0-15) / row r+8 (lanes 16-31),
//     N = lane&15  -> epilogue reduces across each 16-lane half.
// ---------------------------------------------------------------------------
__global__ __launch_bounds__(256) void attn_logits_kernel(
    const float* __restrict__ x, const _Float16* __restrict__ w1t,
    const float* __restrict__ b1, const float* __restrict__ w2,
    const float* __restrict__ b2v, float* __restrict__ wOut) {
  const int lane = threadIdx.x & 31;
  const int wave = threadIdx.x >> 5;
  const int tile = blockIdx.x * 8 + wave;
  if (tile >= NUM_TILES) return;

  const int l15   = lane & 15;
  const int halfw = lane >> 4;
  const int kbase = halfw * 8;

  int koff[8];
#pragma unroll
  for (int p = 0; p < 8; ++p)
    koff[p] = (p < 4) ? (kbase + 2 * p) : (kbase + 8 + 2 * p);

  v8f acc[8];
  v8f zero = {};
#pragma unroll
  for (int nt = 0; nt < 8; ++nt) acc[nt] = zero;

  const float* xrow = x + (size_t)(tile * 16 + l15) * HIDDEN;
  const _Float16* bcol = w1t + (size_t)l15 * HIDDEN;

  for (int kt = 0; kt < 8; ++kt) {
    const float* xk = xrow + kt * 32;
    v16h a;
#pragma unroll
    for (int p = 0; p < 8; ++p) {
      float2 f = *reinterpret_cast<const float2*>(xk + koff[p]);
      a[2 * p]     = (_Float16)f.x;     // f32 -> f16 convert on load
      a[2 * p + 1] = (_Float16)f.y;
    }
#pragma unroll
    for (int nt = 0; nt < 8; ++nt) {
      const _Float16* bk = bcol + (size_t)nt * 16 * HIDDEN + kt * 32;
      v16h b;
#pragma unroll
      for (int p = 0; p < 8; ++p) {
        b[2 * p]     = bk[koff[p]];
        b[2 * p + 1] = bk[koff[p] + 1];
      }
      acc[nt] = __builtin_amdgcn_wmma_f32_16x16x32_f16(
          false, a, false, b, (short)0, acc[nt], false, false);
    }
  }

  // Fused epilogue: +b1, ReLU, dot with W2 column, reduce over the 16 lanes
  // that hold one row's N-slice, then +b2 and sigmoid.
  float part[8];
#pragma unroll
  for (int r = 0; r < 8; ++r) part[r] = 0.0f;
#pragma unroll
  for (int nt = 0; nt < 8; ++nt) {
    int n = nt * 16 + l15;
    float b1v = b1[n];
    float w2v = w2[n];
#pragma unroll
    for (int r = 0; r < 8; ++r) {
      float h = acc[nt][r] + b1v;
      h = h > 0.0f ? h : 0.0f;
      part[r] = fmaf(h, w2v, part[r]);
    }
  }
  for (int mask = 8; mask; mask >>= 1) {
#pragma unroll
    for (int r = 0; r < 8; ++r)
      part[r] += __shfl_xor(part[r], mask, 32);
  }
  if (l15 == 0) {
    float bb = b2v[0];
    int rowBase = tile * 16 + halfw * 8;
#pragma unroll
    for (int r = 0; r < 8; ++r) {
      float z = part[r] + bb;
      wOut[rowBase + r] = 1.0f / (1.0f + __expf(-z));
    }
  }
}

// ---------------------------------------------------------------------------
// Segment max of positive floats via int atomicMax on the bit pattern
// (valid: sigmoid output > 0, gmax init 0). Wave-uniform fast path: batch is
// sorted, so almost all waves see one graph -> one atomic per wave.
// ---------------------------------------------------------------------------
__global__ void seg_max_kernel(const float* __restrict__ wv,
                               const long long* __restrict__ batch,
                               float* __restrict__ gmax) {
  int i  = blockIdx.x * blockDim.x + threadIdx.x;
  int ic = i < NUM_NODES ? i : NUM_NODES - 1;   // clamp tail (dup ok for max)
  int b  = (int)batch[ic];
  float v = wv[ic];
  int b0 = __shfl(b, 0, 32);
  if (__all(b == b0)) {
    float m = v;
    for (int mask = 16; mask; mask >>= 1)
      m = fmaxf(m, __shfl_xor(m, mask, 32));
    if ((threadIdx.x & 31) == 0)
      atomicMax((int*)&gmax[b0], __float_as_int(m));
  } else {
    atomicMax((int*)&gmax[b], __float_as_int(v));
  }
}

__global__ void seg_expsum_kernel(float* __restrict__ wv,   // in: w, out: e
                                  const long long* __restrict__ batch,
                                  const float* __restrict__ gmax,
                                  float* __restrict__ gsum) {
  int i = blockIdx.x * blockDim.x + threadIdx.x;
  bool valid = i < NUM_NODES;
  int ic = valid ? i : NUM_NODES - 1;
  int b  = (int)batch[ic];
  float e = valid ? __expf(wv[ic] - gmax[b]) : 0.0f;
  if (valid) wv[ic] = e;                        // in-place: logits -> exp
  int b0 = __shfl(b, 0, 32);
  if (__all(b == b0)) {
    float s = e;
    for (int mask = 16; mask; mask >>= 1)
      s += __shfl_xor(s, mask, 32);
    if ((threadIdx.x & 31) == 0) atomicAdd(&gsum[b0], s);
  } else {
    atomicAdd(&gsum[b], e);                     // invalid lanes add 0
  }
}

// batch sorted -> start offsets per graph; start[NUM_GRAPHS] = N.
__global__ void seg_bounds_kernel(const long long* __restrict__ batch,
                                  int* __restrict__ gstart) {
  int i = blockIdx.x * blockDim.x + threadIdx.x;
  if (i >= NUM_NODES) return;
  int bi = (int)batch[i];
  int bprev = (i == 0) ? -1 : (int)batch[i - 1];
  for (int g = bprev + 1; g <= bi; ++g) gstart[g] = i;
  if (i == NUM_NODES - 1)
    for (int g = bi + 1; g <= NUM_GRAPHS; ++g) gstart[g] = NUM_NODES;
}

// ---------------------------------------------------------------------------
// out[g,:] = (sum_n e[n]*x[n,:]) / gsum[g], one block per graph, one thread
// per feature. Coalesced streaming of x (second and last HBM pass), no
// atomics. 4-way unrolled + prefetch ahead.
// ---------------------------------------------------------------------------
__global__ __launch_bounds__(256) void readout_kernel(
    const float* __restrict__ x, const float* __restrict__ ev,
    const int* __restrict__ gstart, const float* __restrict__ gsum,
    float* __restrict__ out) {
  int g = blockIdx.x;
  int f = threadIdx.x;
  int s   = gstart[g];
  int end = gstart[g + 1];
  float a0 = 0.f, a1 = 0.f, a2 = 0.f, a3 = 0.f;
  int n = s;
  for (; n + 3 < end; n += 4) {
    __builtin_prefetch(&x[(size_t)(n + 16) * HIDDEN + f], 0, 0);
    float p0 = ev[n], p1 = ev[n + 1], p2 = ev[n + 2], p3 = ev[n + 3];
    a0 = fmaf(x[(size_t)(n + 0) * HIDDEN + f], p0, a0);
    a1 = fmaf(x[(size_t)(n + 1) * HIDDEN + f], p1, a1);
    a2 = fmaf(x[(size_t)(n + 2) * HIDDEN + f], p2, a2);
    a3 = fmaf(x[(size_t)(n + 3) * HIDDEN + f], p3, a3);
  }
  for (; n < end; ++n)
    a0 = fmaf(x[(size_t)n * HIDDEN + f], ev[n], a0);
  float total = (a0 + a1) + (a2 + a3);
  float denom = gsum[g];
  out[(size_t)g * HIDDEN + f] = (end > s && denom > 0.0f) ? total / denom : 0.0f;
}

// ---------------------------------------------------------------------------
extern "C" void kernel_launch(void* const* d_in, const int* in_sizes, int n_in,
                              void* d_out, int out_size, void* d_ws,
                              size_t ws_size, hipStream_t stream) {
  const float* x      = (const float*)d_in[0];
  const float* W1     = (const float*)d_in[1];
  const float* b1     = (const float*)d_in[2];
  const float* W2     = (const float*)d_in[3];
  const float* b2     = (const float*)d_in[4];
  const long long* bt = (const long long*)d_in[5];
  float* out = (float*)d_out;

  // Workspace carve (~4.1 MB total): e/logits, gmax, gsum, bounds, W1T(f16).
  char* ws = (char*)d_ws;
  size_t off = 0;
  auto carve = [&](size_t bytes) -> void* {
    void* p = ws + off;
    off += (bytes + 255) & ~(size_t)255;
    return p;
  };
  float*     wv     = (float*)carve((size_t)NUM_NODES * 4);
  float*     gmax   = (float*)carve((size_t)NUM_GRAPHS * 4);
  float*     gsum   = (float*)carve((size_t)NUM_GRAPHS * 4);
  int*       gstart = (int*)carve((size_t)(NUM_GRAPHS + 1) * 4);
  _Float16*  w1t    = (_Float16*)carve((size_t)RH * HIDDEN * 2);

  prep_w1t_kernel<<<(RH * HIDDEN) / 256, 256, 0, stream>>>(W1, w1t);
  init_stats_kernel<<<NUM_GRAPHS / 256, 256, 0, stream>>>(gmax, gsum);

  // 62500 sixteen-node tiles, 8 waves (tiles) per 256-thread block.
  attn_logits_kernel<<<(NUM_TILES + 7) / 8, 256, 0, stream>>>(
      x, w1t, b1, W2, b2, wv);

  int nodeBlocks = (NUM_NODES + 255) / 256;
  seg_max_kernel<<<nodeBlocks, 256, 0, stream>>>(wv, bt, gmax);
  seg_expsum_kernel<<<nodeBlocks, 256, 0, stream>>>(wv, bt, gmax, gsum);
  seg_bounds_kernel<<<nodeBlocks, 256, 0, stream>>>(bt, gstart);

  readout_kernel<<<NUM_GRAPHS, 256, 0, stream>>>(x, wv, gstart, gsum, out);
}
